// HybridTokenizer_49211735277865
// MI455X (gfx1250) — matrix operand order repaired and checked by reading, not compiled, not run on hardware
//
#include <hip/hip_runtime.h>

// ---------------- problem constants ----------------
constexpr int Bn = 8;          // batch
constexpr int Kp = 8192;       // points
constexpr int Cn = 256;        // channels
constexpr int En = 256;        // embed dim
constexpr int Ln = 256;        // NUM_LEARNED
constexpr int NG = 256;        // NUM_GRID
constexpr int Tn = 512;        // NUM_TOKENS
constexpr int Hn = 8;          // heads
constexpr int Gn = 7;          // grid resolution

#define DEVFN static __device__ __forceinline__

typedef __attribute__((ext_vector_type(16))) __bf16 v16bf;
typedef __attribute__((ext_vector_type(8)))  float  v8f;
typedef __attribute__((ext_vector_type(4)))  unsigned int u32x4;
typedef __attribute__((ext_vector_type(8)))  int i32x8;
typedef __attribute__((ext_vector_type(4)))  int i32x4;

DEVFN v8f vzero8() {
  v8f z;
#pragma unroll
  for (int i = 0; i < 8; i++) z[i] = 0.0f;
  return z;
}

DEVFN v8f wmma_bf16(v16bf a, v16bf b, v8f c) {
  return __builtin_amdgcn_wmma_f32_16x16x32_bf16(false, a, false, b,
                                                 (short)0, c, false, false);
}

// A-fragment (16x32, bf16): lane holds row = lane&15; element i -> k index:
DEVFN int a_kidx(int i, int hi) { return hi * 16 + i; }

// 16-lane-group reductions (rows of a C/D tile live in one half-wave)
DEVFN float grp_max(float v) {
  v = fmaxf(v, __shfl_xor(v, 1, 32));
  v = fmaxf(v, __shfl_xor(v, 2, 32));
  v = fmaxf(v, __shfl_xor(v, 4, 32));
  v = fmaxf(v, __shfl_xor(v, 8, 32));
  return v;
}
DEVFN float grp_sum(float v) {
  v += __shfl_xor(v, 1, 32);
  v += __shfl_xor(v, 2, 32);
  v += __shfl_xor(v, 4, 32);
  v += __shfl_xor(v, 8, 32);
  return v;
}

// ---- Tensor Data Mover: 2D tile (tile_d1 rows x tile_d0 elems) -> LDS -----
// D# layout per CDNA5 ISA (group0 + group1, exact-fit
// tensor dims so no OOB path is exercised; groups 2/3 zero => <=2D tensor).
// data_size_code: 0=1B 1=2B 2=4B 3=8B. stride in elements.
// This toolchain declares the 6-arg builtin form:
//   (u32x4 g0, i32x8 g1, i32x4 g2, i32x4 g3, i32x8 extra, i32 cpol)
DEVFN void tdm_load_2d(void* lds_ptr, const void* gptr,
                       unsigned data_size_code, unsigned tile_d0,
                       unsigned tile_d1, unsigned stride_elems) {
  unsigned long long ga = (unsigned long long)gptr;
  unsigned lds = (unsigned)(unsigned long long)lds_ptr;  // flat low 32 = LDS addr
  u32x4 g0;
  g0[0] = 1u;                                        // count=1, user descriptor
  g0[1] = lds;                                       // lds_addr [63:32]
  g0[2] = (unsigned)(ga & 0xFFFFFFFFu);              // global_addr [95:64]
  g0[3] = (unsigned)((ga >> 32) & 0x1FFFFFFu) | (2u << 30);  // addr[56:32]|type=2
  i32x8 g1;
  g1[0] = (int)(data_size_code << 16);               // mask=0, data_size
  g1[1] = (int)((tile_d0 & 0xFFFFu) << 16);          // tensor_dim0 lo16 @bit48
  g1[2] = (int)((tile_d0 >> 16) | ((tile_d1 & 0xFFFFu) << 16)); // d0 hi|d1 lo
  g1[3] = (int)((tile_d1 >> 16) | ((tile_d0 & 0xFFFFu) << 16)); // d1 hi|tile_dim0
  g1[4] = (int)(tile_d1 & 0xFFFFu);                  // tile_dim1 (tile_dim2=0)
  g1[5] = (int)stride_elems;                         // tensor_dim0_stride lo32
  g1[6] = 0;
  g1[7] = 0;
  i32x4 z4;
  z4[0] = z4[1] = z4[2] = z4[3] = 0;
  i32x8 z8;
#pragma unroll
  for (int i = 0; i < 8; i++) z8[i] = 0;
  __builtin_amdgcn_tensor_load_to_lds(g0, g1, z4, z4, z8, 0);
}

// ---------------- workspace layout (bytes) ----------------
constexpr size_t OFF_MINMAX = 0;                               // B*6*4 -> 256
constexpr size_t OFF_CELL   = 256;                             // B*K*4
constexpr size_t AGG_XYZ    = OFF_CELL + (size_t)Bn * Kp * 4;  // B*NG*3*4
constexpr size_t AGG_CNT    = AGG_XYZ + (size_t)Bn * NG * 3 * 4;
constexpr size_t AGG_FEAT   = AGG_CNT + (size_t)Bn * NG * 4;   // B*NG*C*4
constexpr size_t OFF_WKV    = AGG_FEAT + (size_t)Bn * NG * Cn * 4; // 768*256*2
constexpr size_t OFF_WOUT   = OFF_WKV + (size_t)768 * En * 2;
constexpr size_t OFF_QS     = OFF_WOUT + (size_t)En * En * 2;
constexpr size_t OFF_KB     = OFF_QS + (size_t)Ln * En * 2;
constexpr size_t OFF_VB     = OFF_KB + (size_t)Bn * Kp * En * 2;
constexpr size_t OFF_CTX    = OFF_VB + (size_t)Bn * Kp * En * 2;
constexpr size_t OFF_AXYZ   = OFF_CTX + (size_t)Bn * Ln * En * 4;
constexpr size_t OFF_X      = OFF_AXYZ + (size_t)Bn * Hn * Ln * 3 * 4;
constexpr size_t AGG_FLOATS = (size_t)Bn * NG * (3 + 1 + Cn);

// ---------------- small utility kernels ----------------
__global__ __launch_bounds__(256) void ht_zero_kernel(float* p, size_t n) {
  size_t i = (size_t)blockIdx.x * 256 + threadIdx.x;
  if (i < n) p[i] = 0.0f;
}

__global__ __launch_bounds__(768) void ht_convw_kernel(
    const float* __restrict__ ipw, const float* __restrict__ opw,
    __bf16* __restrict__ Wkv, __bf16* __restrict__ Wout) {
  size_t i = (size_t)blockIdx.x * blockDim.x + threadIdx.x;
  if (i < (size_t)768 * En) Wkv[i] = (__bf16)ipw[i];
  if (i < (size_t)En * En)  Wout[i] = (__bf16)opw[i];
}

// ---------------- grid aggregation ----------------
__global__ __launch_bounds__(256) void ht_minmax_kernel(
    const float* __restrict__ xyz, float* __restrict__ mm) {
  int b = blockIdx.x, tid = threadIdx.x;
  float mn[3], mx[3];
#pragma unroll
  for (int j = 0; j < 3; j++) { mn[j] = 3.0e38f; mx[j] = -3.0e38f; }
  for (int k = tid; k < Kp; k += 256) {
    const float* p = &xyz[((size_t)b * Kp + k) * 3];
#pragma unroll
    for (int j = 0; j < 3; j++) {
      mn[j] = fminf(mn[j], p[j]);
      mx[j] = fmaxf(mx[j], p[j]);
    }
  }
  __shared__ float s[256];
#pragma unroll
  for (int j = 0; j < 3; j++) {
    s[tid] = mn[j]; __syncthreads();
    for (int st = 128; st > 0; st >>= 1) {
      if (tid < st) s[tid] = fminf(s[tid], s[tid + st]);
      __syncthreads();
    }
    if (tid == 0) mm[b * 6 + j] = s[0];
    __syncthreads();
    s[tid] = mx[j]; __syncthreads();
    for (int st = 128; st > 0; st >>= 1) {
      if (tid < st) s[tid] = fmaxf(s[tid], s[tid + st]);
      __syncthreads();
    }
    if (tid == 0) mm[b * 6 + 3 + j] = s[0];
    __syncthreads();
  }
}

__global__ __launch_bounds__(256) void ht_cell_kernel(
    const float* __restrict__ xyz, const float* __restrict__ mm,
    int* __restrict__ cellidx, float* __restrict__ aggxyz,
    float* __restrict__ cnt) {
  int idx = blockIdx.x * 256 + threadIdx.x;   // b*K + k
  if (idx >= Bn * Kp) return;
  int b = idx >> 13;
  float p[3];
#pragma unroll
  for (int j = 0; j < 3; j++) p[j] = xyz[(size_t)idx * 3 + j];
  bool valid = (fabsf(p[0]) + fabsf(p[1]) + fabsf(p[2])) > 0.0f;
  int gi[3];
#pragma unroll
  for (int j = 0; j < 3; j++) {
    float mn = mm[b * 6 + j], mx = mm[b * 6 + 3 + j];
    float nrm = (p[j] - mn) / (mx - mn + 1e-6f);
    int g = (int)(nrm * (float)Gn);
    gi[j] = min(max(g, 0), Gn - 1);
  }
  int cell = gi[0] * (Gn * Gn) + gi[1] * Gn + gi[2];
  int store = (valid && cell < NG) ? cell : -1;   // cells >= NG are sliced away
  cellidx[idx] = store;
  if (store >= 0) {
    float* ax = &aggxyz[((size_t)b * NG + cell) * 3];
    atomicAdd(&ax[0], p[0]);
    atomicAdd(&ax[1], p[1]);
    atomicAdd(&ax[2], p[2]);
    atomicAdd(&cnt[b * NG + cell], 1.0f);
  }
}

__global__ __launch_bounds__(256) void ht_featscat_kernel(
    const float* __restrict__ features, const int* __restrict__ cellidx,
    float* __restrict__ aggfeat) {
  int idx = blockIdx.x;              // b*K + k
  int cell = cellidx[idx];
  if (cell < 0) return;
  int b = idx >> 13, k = idx & (Kp - 1);
  int c = threadIdx.x;
  atomicAdd(&aggfeat[((size_t)b * NG + cell) * Cn + c],
            features[((size_t)b * Cn + c) * Kp + k]);
}

__global__ __launch_bounds__(256) void ht_gridfin_kernel(
    const float* __restrict__ aggxyz, const float* __restrict__ aggfeat,
    const float* __restrict__ cnt, float* __restrict__ out) {
  int t = blockIdx.x;                // b*NG + cell
  int b = t >> 8, cell = t & 255;
  float inv = 1.0f / fmaxf(cnt[t], 1.0f);
  float* out_xyz = out;
  float* out_feat = out + (size_t)Bn * Tn * 3;
  if (threadIdx.x < 3)
    out_xyz[((size_t)b * Tn + cell) * 3 + threadIdx.x] =
        aggxyz[(size_t)t * 3 + threadIdx.x] * inv;
  int c = threadIdx.x;
  out_feat[((size_t)b * Cn + c) * Tn + cell] = aggfeat[(size_t)t * Cn + c] * inv;
}

// ---------------- Q projection: Qs[l][f] = bf16((qt·wq^T + bq) * D^-0.5) ----
__global__ __launch_bounds__(32) void ht_qproj_kernel(
    const float* __restrict__ qt, const __bf16* __restrict__ Wkv,
    const float* __restrict__ ipb, __bf16* __restrict__ Qs) {
  int lt = blockIdx.x, ft = blockIdx.y;
  int lane = threadIdx.x, col = lane & 15, hi = lane >> 4;
  v8f acc = vzero8();
  for (int e0 = 0; e0 < En; e0 += 32) {
    v16bf a, bw;
#pragma unroll
    for (int i = 0; i < 16; i++) {
      a[i] = (__bf16)qt[(size_t)(lt * 16 + col) * En + e0 + a_kidx(i, hi)];
      bw[i] = Wkv[(size_t)(ft * 16 + col) * En + e0 + hi * 16 + i];
    }
    acc = wmma_bf16(a, bw, acc);
  }
  float bias = ipb[ft * 16 + col];
  const float scl = 0.17677669529663689f;   // 32^-0.5
#pragma unroll
  for (int r = 0; r < 8; r++)
    Qs[(size_t)(lt * 16 + r + 8 * hi) * En + ft * 16 + col] =
        (__bf16)((acc[r] + bias) * scl);
}

// --------- K/V projection: features^T · w^T + b -> bf16 (TDM-staged) -------
__global__ __launch_bounds__(256) void ht_kvproj_kernel(
    const float* __restrict__ features, const __bf16* __restrict__ Wkv,
    const float* __restrict__ ipb, __bf16* __restrict__ Kb,
    __bf16* __restrict__ Vb) {
  __shared__ float As[2][32][16];   // [buf][e][k-col], TDM tile
  int b = blockIdx.y;
  int k0 = blockIdx.x * 16;
  int tid = threadIdx.x, wave = tid >> 5, lane = tid & 31;
  int col = lane & 15, hi = lane >> 4;
  int f0 = wave * 32;               // this wave's 32 output columns
  v8f ak0 = vzero8(), ak1 = vzero8(), av0 = vzero8(), av1 = vzero8();

  const float* Fbase = features + (size_t)b * Cn * Kp + k0;
  constexpr int NCH = En / 32;      // 8 chunks
  if (wave == 0)
    tdm_load_2d(&As[0][0][0], Fbase, 2u, 16u, 32u, (unsigned)Kp);

  for (int ci = 0; ci < NCH; ci++) {
    int buf = ci & 1;
    int e0 = ci * 32;
    if (wave == 0) {
      if (ci + 1 < NCH) {
        tdm_load_2d(&As[buf ^ 1][0][0], Fbase + (size_t)(e0 + 32) * Kp,
                    2u, 16u, 32u, (unsigned)Kp);
        __builtin_amdgcn_s_wait_tensorcnt(1);   // in-order: chunk ci landed
      } else {
        __builtin_amdgcn_s_wait_tensorcnt(0);
      }
    }
    __syncthreads();
    v16bf af, bk0, bk1, bv0, bv1;
#pragma unroll
    for (int i = 0; i < 16; i++) {
      af[i] = (__bf16)As[buf][a_kidx(i, hi)][col];
      int e = e0 + hi * 16 + i;
      bk0[i] = Wkv[(size_t)(256 + f0 + col) * En + e];
      bk1[i] = Wkv[(size_t)(256 + f0 + 16 + col) * En + e];
      bv0[i] = Wkv[(size_t)(512 + f0 + col) * En + e];
      bv1[i] = Wkv[(size_t)(512 + f0 + 16 + col) * En + e];
    }
    ak0 = wmma_bf16(af, bk0, ak0);
    ak1 = wmma_bf16(af, bk1, ak1);
    av0 = wmma_bf16(af, bv0, av0);
    av1 = wmma_bf16(af, bv1, av1);
    __syncthreads();                 // done with buf before TDM reuses it
  }
  float bk_c0 = ipb[256 + f0 + col], bk_c1 = ipb[256 + f0 + 16 + col];
  float bv_c0 = ipb[512 + f0 + col], bv_c1 = ipb[512 + f0 + 16 + col];
#pragma unroll
  for (int r = 0; r < 8; r++) {
    size_t row = (size_t)b * Kp + k0 + r + 8 * hi;
    Kb[row * En + f0 + col]      = (__bf16)(ak0[r] + bk_c0);
    Kb[row * En + f0 + 16 + col] = (__bf16)(ak1[r] + bk_c1);
    Vb[row * En + f0 + col]      = (__bf16)(av0[r] + bv_c0);
    Vb[row * En + f0 + 16 + col] = (__bf16)(av1[r] + bv_c1);
  }
}

// -------- flash attention: one workgroup per (b,h); 16 waves share K/V -----
__global__ __launch_bounds__(512) void ht_attn_kernel(
    const __bf16* __restrict__ Qs, const __bf16* __restrict__ Kb,
    const __bf16* __restrict__ Vb, const float* __restrict__ xyz,
    float* __restrict__ ctx, float* __restrict__ axyz) {
  __shared__ __bf16 Ks[2][32][32];      // [buf][key][d]  (TDM tile)
  __shared__ __bf16 Vs[2][32][32];      // [buf][key][d]  (TDM tile)
  __shared__ __bf16 Ps[16][16][32];     // per-wave P tile [l][k]
  int h = blockIdx.x, b = blockIdx.y;
  int tid = threadIdx.x, wave = tid >> 5, lane = tid & 31;
  int col = lane & 15, hi = lane >> 4;
  int l0 = wave * 16;                   // each wave owns one l-tile

  v16bf qa;
#pragma unroll
  for (int i = 0; i < 16; i++)
    qa[i] = Qs[(size_t)(l0 + col) * En + h * 32 + a_kidx(i, hi)];

  v8f acc0 = vzero8(), acc1 = vzero8();
  float m_run[8], denom[8], px[8][3];
#pragma unroll
  for (int r = 0; r < 8; r++) {
    m_run[r] = -3.0e38f; denom[r] = 0.0f;
    px[r][0] = px[r][1] = px[r][2] = 0.0f;
  }

  const __bf16* Kbase = Kb + (size_t)b * Kp * En + h * 32;
  const __bf16* Vbase = Vb + (size_t)b * Kp * En + h * 32;
  constexpr int NCH = Kp / 32;          // 256 chunks
  if (wave == 0) {
    tdm_load_2d(&Ks[0][0][0], Kbase, 1u, 32u, 32u, (unsigned)En);
    tdm_load_2d(&Vs[0][0][0], Vbase, 1u, 32u, 32u, (unsigned)En);
  }

  for (int ci = 0; ci < NCH; ci++) {
    int buf = ci & 1;
    int k0 = ci * 32;
    if (wave == 0) {
      if (ci + 1 < NCH) {
        tdm_load_2d(&Ks[buf ^ 1][0][0], Kbase + (size_t)(k0 + 32) * En,
                    1u, 32u, 32u, (unsigned)En);
        tdm_load_2d(&Vs[buf ^ 1][0][0], Vbase + (size_t)(k0 + 32) * En,
                    1u, 32u, 32u, (unsigned)En);
        __builtin_amdgcn_s_wait_tensorcnt(2);   // in-order: chunk ci landed
      } else {
        __builtin_amdgcn_s_wait_tensorcnt(0);
      }
    }
    __syncthreads();

    // S = Q (16l x 32d) x K^T : two 16-key halves from LDS
    v16bf kb0, kb1;
#pragma unroll
    for (int i = 0; i < 16; i++) {
      kb0[i] = Ks[buf][col][hi * 16 + i];
      kb1[i] = Ks[buf][16 + col][hi * 16 + i];
    }
    v8f s0 = wmma_bf16(qa, kb0, vzero8());
    v8f s1 = wmma_bf16(qa, kb1, vzero8());

    float x0[3], x1[3];
#pragma unroll
    for (int j = 0; j < 3; j++) {
      x0[j] = xyz[((size_t)b * Kp + k0 + col) * 3 + j];
      x1[j] = xyz[((size_t)b * Kp + k0 + 16 + col) * 3 + j];
    }

#pragma unroll
    for (int r = 0; r < 8; r++) {
      float v0 = s0[r], v1 = s1[r];
      float mloc = grp_max(fmaxf(v0, v1));
      float mnew = fmaxf(m_run[r], mloc);
      float corr = __expf(m_run[r] - mnew);
      m_run[r] = mnew;
      float p0 = __expf(v0 - mnew), p1 = __expf(v1 - mnew);
      denom[r] = denom[r] * corr + grp_sum(p0 + p1);
      Ps[wave][r + 8 * hi][col] = (__bf16)p0;
      Ps[wave][r + 8 * hi][16 + col] = (__bf16)p1;
      acc0[r] *= corr;
      acc1[r] *= corr;
#pragma unroll
      for (int j = 0; j < 3; j++)
        px[r][j] = px[r][j] * corr + p0 * x0[j] + p1 * x1[j];
    }
    __syncthreads();

    // ctx += P (16l x 32k) x V (32k x 32d) from LDS
    v16bf pa, vb0, vb1;
#pragma unroll
    for (int i = 0; i < 16; i++) {
      pa[i] = Ps[wave][col][a_kidx(i, hi)];
      vb0[i] = Vs[buf][hi * 16 + i][col];
      vb1[i] = Vs[buf][hi * 16 + i][16 + col];
    }
    acc0 = wmma_bf16(pa, vb0, acc0);
    acc1 = wmma_bf16(pa, vb1, acc1);
    __syncthreads();                    // done with buf before TDM reuses it
  }

#pragma unroll
  for (int r = 0; r < 8; r++) {
    float inv = 1.0f / denom[r];
    size_t row = (size_t)b * Ln + l0 + r + 8 * hi;
    ctx[row * En + h * 32 + col] = acc0[r] * inv;
    ctx[row * En + h * 32 + 16 + col] = acc1[r] * inv;
#pragma unroll
    for (int j = 0; j < 3; j++) {
      float v = grp_sum(px[r][j]);
      if (col == 0)
        axyz[(((size_t)b * Hn + h) * Ln + l0 + r + 8 * hi) * 3 + j] = v * inv;
    }
  }
}

// ---------------- out-proj + residual: X = q + ctx·Wout^T + bout -----------
__global__ __launch_bounds__(32) void ht_outproj_kernel(
    const float* __restrict__ ctx, const __bf16* __restrict__ Wout,
    const float* __restrict__ opb, const float* __restrict__ qt,
    float* __restrict__ Xb) {
  int row0 = blockIdx.x * 16;     // over B*L rows (tiles never cross batches)
  int ft = blockIdx.y;
  int lane = threadIdx.x, col = lane & 15, hi = lane >> 4;
  v8f acc = vzero8();
  for (int e0 = 0; e0 < En; e0 += 32) {
    v16bf a, bw;
#pragma unroll
    for (int i = 0; i < 16; i++) {
      a[i] = (__bf16)ctx[(size_t)(row0 + col) * En + e0 + a_kidx(i, hi)];
      bw[i] = Wout[(size_t)(ft * 16 + col) * En + e0 + hi * 16 + i];
    }
    acc = wmma_bf16(a, bw, acc);
  }
  int f = ft * 16 + col;
  float bias = opb[f];
#pragma unroll
  for (int r = 0; r < 8; r++) {
    int row = row0 + r + 8 * hi;
    int l = row & (Ln - 1);
    Xb[(size_t)row * En + f] = qt[(size_t)l * En + f] + acc[r] + bias;
  }
}

// ---------------- layernorm -> out_feat[:, :, NG+l] ------------------------
__global__ __launch_bounds__(256) void ht_ln_kernel(
    const float* __restrict__ Xb, const float* __restrict__ lnw,
    const float* __restrict__ lnb, float* __restrict__ out_feat) {
  __shared__ float s1[256], s2[256];
  int row = blockIdx.x;           // b*L + l
  int c = threadIdx.x;
  float v = Xb[(size_t)row * En + c];
  s1[c] = v; s2[c] = v * v;
  __syncthreads();
  for (int st = 128; st > 0; st >>= 1) {
    if (c < st) { s1[c] += s1[c + st]; s2[c] += s2[c + st]; }
    __syncthreads();
  }
  float mu = s1[0] * (1.0f / En);
  float var = s2[0] * (1.0f / En) - mu * mu;
  float y = (v - mu) * rsqrtf(var + 1e-5f) * lnw[c] + lnb[c];
  int b = row >> 8, l = row & 255;
  out_feat[((size_t)b * Cn + c) * Tn + NG + l] = y;
}

// ---------------- learned token xyz: mean over heads -----------------------
__global__ __launch_bounds__(256) void ht_xyzout_kernel(
    const float* __restrict__ axyz, float* __restrict__ out) {
  int idx = blockIdx.x * 256 + threadIdx.x;   // b*L*3
  if (idx >= Bn * Ln * 3) return;
  int b = idx / (Ln * 3);
  int rem = idx - b * (Ln * 3);
  int l = rem / 3, j = rem - l * 3;
  float s = 0.0f;
#pragma unroll
  for (int h = 0; h < Hn; h++)
    s += axyz[(((size_t)b * Hn + h) * Ln + l) * 3 + j];
  out[((size_t)b * Tn + NG + l) * 3 + j] = s * (1.0f / Hn);
}

// ---------------- launcher ----------------
extern "C" void kernel_launch(void* const* d_in, const int* in_sizes, int n_in,
                              void* d_out, int out_size, void* d_ws,
                              size_t ws_size, hipStream_t stream) {
  (void)in_sizes; (void)n_in; (void)out_size; (void)ws_size;
  const float* xyz  = (const float*)d_in[0];
  const float* feat = (const float*)d_in[1];
  const float* qt   = (const float*)d_in[2];
  const float* ipw  = (const float*)d_in[3];
  const float* ipb  = (const float*)d_in[4];
  const float* opw  = (const float*)d_in[5];
  const float* opb  = (const float*)d_in[6];
  const float* lnw  = (const float*)d_in[7];
  const float* lnb  = (const float*)d_in[8];
  char* ws = (char*)d_ws;
  float* out = (float*)d_out;

  float*  mm      = (float*)(ws + OFF_MINMAX);
  int*    cell    = (int*)(ws + OFF_CELL);
  float*  aggxyz  = (float*)(ws + AGG_XYZ);
  float*  cnt     = (float*)(ws + AGG_CNT);
  float*  aggfeat = (float*)(ws + AGG_FEAT);
  __bf16* Wkv     = (__bf16*)(ws + OFF_WKV);
  __bf16* Wout    = (__bf16*)(ws + OFF_WOUT);
  __bf16* Qsb     = (__bf16*)(ws + OFF_QS);
  __bf16* Kb      = (__bf16*)(ws + OFF_KB);
  __bf16* Vb      = (__bf16*)(ws + OFF_VB);
  float*  ctx     = (float*)(ws + OFF_CTX);
  float*  axyz    = (float*)(ws + OFF_AXYZ);
  float*  Xb      = (float*)(ws + OFF_X);

  // grid aggregation path
  ht_zero_kernel<<<(unsigned)((AGG_FLOATS + 255) / 256), 256, 0, stream>>>(
      aggxyz, AGG_FLOATS);
  ht_minmax_kernel<<<Bn, 256, 0, stream>>>(xyz, mm);
  ht_cell_kernel<<<(Bn * Kp) / 256, 256, 0, stream>>>(xyz, mm, cell, aggxyz, cnt);
  ht_featscat_kernel<<<Bn * Kp, 256, 0, stream>>>(feat, cell, aggfeat);
  ht_gridfin_kernel<<<Bn * NG, 256, 0, stream>>>(aggxyz, aggfeat, cnt, out);

  // learned-token attention path
  ht_convw_kernel<<<(768 * En) / 768, 768, 0, stream>>>(ipw, opw, Wkv, Wout);
  ht_qproj_kernel<<<dim3(Ln / 16, En / 16), 32, 0, stream>>>(qt, Wkv, ipb, Qsb);
  ht_kvproj_kernel<<<dim3(Kp / 16, Bn), 256, 0, stream>>>(feat, Wkv, ipb, Kb, Vb);
  ht_attn_kernel<<<dim3(Hn, Bn), 512, 0, stream>>>(Qsb, Kb, Vb, xyz, ctx, axyz);
  ht_outproj_kernel<<<dim3((Bn * Ln) / 16, En / 16), 32, 0, stream>>>(
      ctx, Wout, opb, qt, Xb);
  ht_ln_kernel<<<Bn * Ln, 256, 0, stream>>>(Xb, lnw, lnb,
                                            out + (size_t)Bn * Tn * 3);
  ht_xyzout_kernel<<<(Bn * Ln * 3 + 255) / 256, 256, 0, stream>>>(axyz, out);
}
